// OptimizedMultiheadAttention_62036507623685
// MI455X (gfx1250) — matrix operand order repaired and checked by reading, not compiled, not run on hardware
//
#include <hip/hip_runtime.h>

// Problem constants (fixed by the reference).
#define B_   2
#define S_   2048
#define E_   1024
#define H_   16
#define HD_  64
#define SCALE_ 0.125f   // HD^-0.5

// gfx1250 async global->LDS path (ASYNCcnt), with safe fallback.
#if defined(__AMDGCN__) && \
    __has_builtin(__builtin_amdgcn_global_load_async_to_lds_b128) && \
    __has_builtin(__builtin_amdgcn_s_wait_asynccnt)
#define ASYNC_LDS 1
#else
#define ASYNC_LDS 0
#endif

#if ASYNC_LDS
typedef int v4i_async __attribute__((vector_size(16)));
#endif

// ---------- CDNA5 WMMA types ----------
typedef __attribute__((ext_vector_type(16))) __bf16 v16bf;
typedef __attribute__((ext_vector_type(8)))  float  v8f;

__device__ __forceinline__ unsigned short f2bf(float f) {
  union { float f; unsigned u; } x; x.f = f;
  unsigned u = x.u + 0x7FFFu + ((x.u >> 16) & 1u);   // round-to-nearest-even
  return (unsigned short)(u >> 16);
}

union FragU { uint4 q[2]; v16bf v; };

__device__ __forceinline__ v16bf load_frag_lds(const unsigned short* p0,
                                               const unsigned short* p1) {
  FragU u;
  u.q[0] = *(const uint4*)p0;
  u.q[1] = *(const uint4*)p1;
  return u.v;
}

// ---------- generic bf16 WMMA GEMM: C[M,N] = A[M,K] * B[K,N] ----------
// A row-major (bf16 in memory, or fp32 converted on stage), B k-major.
// Requires: M % 128 == 0, N % 64 == 0, K % 32 == 0 (true for all call sites).
// Block: 256 threads (8 wave32s). Tile 128x64, K-step 32, double-buffered LDS.
template <bool A_F32, class Epi>
__global__ __launch_bounds__(256)
void gemm_wmma(const void* __restrict__ Abase,
               const unsigned short* __restrict__ Bbase,
               int M, int N, int K, int lda, int ldb,
               long long strideAz, long long strideBz, Epi epi) {
  __shared__ alignas(16) unsigned short As[2][128 * 32];  // [buf][row][k]
  __shared__ alignas(16) unsigned short Bs[2][64 * 32];   // [buf][n][k] (transposed)

  const int t    = threadIdx.x;
  const int lane = t & 31;
  const int wave = t >> 5;
  const int bm   = blockIdx.y * 128;
  const int bn   = blockIdx.x * 64;
  const int z    = blockIdx.z;

  const float*          Af = (const float*)Abase          + (size_t)z * strideAz;
  const unsigned short* Ab = (const unsigned short*)Abase + (size_t)z * strideAz;
  const unsigned short* Bg = Bbase + (size_t)z * strideBz;

  // ---- panel staging ----
  auto stageA = [&](int buf, int k0) {
    if constexpr (A_F32) {
      // fp32 probabilities -> bf16 while staging (P@V consumes d_out directly)
      #pragma unroll
      for (int p = 0; p < 4; ++p) {
        const int r = p * 32 + (t >> 3);
        const int c = (t & 7) * 4;
        const float4 g = *(const float4*)&Af[(size_t)(bm + r) * lda + k0 + c];
        const unsigned lo = (unsigned)f2bf(g.x) | ((unsigned)f2bf(g.y) << 16);
        const unsigned hi = (unsigned)f2bf(g.z) | ((unsigned)f2bf(g.w) << 16);
        *(uint2*)&As[buf][r * 32 + c] = make_uint2(lo, hi);
      }
    } else {
      #pragma unroll
      for (int p = 0; p < 2; ++p) {
        const int r = p * 64 + (t >> 2);
        const int c = (t & 3) * 8;
#if ASYNC_LDS
        // async DMA: 16B/lane global -> LDS, no VGPR bounce (ASYNCcnt)
        __builtin_amdgcn_global_load_async_to_lds_b128(
            (__attribute__((address_space(1))) v4i_async*)
                &Ab[(size_t)(bm + r) * lda + k0 + c],
            (__attribute__((address_space(3))) v4i_async*)
                &As[buf][r * 32 + c],
            0, 0);
#else
        *(uint4*)&As[buf][r * 32 + c] =
            *(const uint4*)&Ab[(size_t)(bm + r) * lda + k0 + c];
#endif
      }
    }
  };
  auto stageB = [&](int buf, int k0) {
    const int k  = t >> 3;
    const int n0 = (t & 7) * 8;
    const uint4 g = *(const uint4*)&Bg[(size_t)(k0 + k) * ldb + bn + n0];
    const unsigned short* sp = (const unsigned short*)&g;
    #pragma unroll
    for (int j = 0; j < 8; ++j) Bs[buf][(n0 + j) * 32 + k] = sp[j];
  };
  auto wait_async = [&]() {
#if ASYNC_LDS
    if constexpr (!A_F32) __builtin_amdgcn_s_wait_asynccnt(0);
#endif
  };

  v8f acc[4] = {};

  // ---- prologue: fill buffer 0 ----
  stageA(0, 0);
  stageB(0, 0);
  wait_async();
  __syncthreads();

  int buf = 0;
  for (int k0 = 0; k0 < K; k0 += 32) {
    // stage next panel into the other buffer while computing this one
    if (k0 + 32 < K) {
      stageA(buf ^ 1, k0 + 32);
      stageB(buf ^ 1, k0 + 32);
      __builtin_prefetch(&Bg[(size_t)(k0 + 64) * ldb + bn], 0, 0);
    }

    // ---- WMMA: wave computes 16x64 strip (4 tiles), A fragment reused ----
    const int rowl = wave * 16 + (lane & 15);
    const int kbA  = (lane >> 4) * 8;   // ISA 16-bit A layout: K {kb..kb+7, kb+16..kb+23}
    const v16bf a = load_frag_lds(&As[buf][rowl * 32 + kbA],
                                  &As[buf][rowl * 32 + kbA + 16]);
    const int kbB = (lane >> 4) * 8;    // B layout: K {kb..kb+7, kb+16..kb+23}, same as A
    #pragma unroll
    for (int nt = 0; nt < 4; ++nt) {
      const int colg = nt * 16 + (lane & 15);
      const v16bf b = load_frag_lds(&Bs[buf][colg * 32 + kbB],
                                    &Bs[buf][colg * 32 + kbB + 16]);
      acc[nt] = __builtin_amdgcn_wmma_f32_16x16x32_bf16(
          false, a, false, b, (short)0, acc[nt], false, false);
    }

    wait_async();       // next panel's async copies must land before the barrier
    __syncthreads();
    buf ^= 1;
  }

  // ---- epilogue: C/D layout -> (m, n) ----
  #pragma unroll
  for (int nt = 0; nt < 4; ++nt) {
    #pragma unroll
    for (int r = 0; r < 8; ++r) {
      const int m = bm + wave * 16 + r + ((lane >> 4) << 3);
      const int n = bn + nt * 16 + (lane & 15);
      epi(z, m, n, acc[nt][r]);
    }
  }
}

// ---------- epilogue functors ----------
struct EpiQKV {   // split fused qkv into q (scaled), k^T, v head-major bf16 buffers
  unsigned short* q; unsigned short* kt; unsigned short* v; const float* bias;
  __device__ void operator()(int, int m, int n, float val) const {
    val += bias[n];
    const int part = n >> 10, e = n & (E_ - 1);
    const int h = e >> 6, d = e & (HD_ - 1);
    const int b = m >> 11, s = m & (S_ - 1);
    const size_t head = (size_t)b * H_ + h;
    if (part == 0)      q [(head * S_ + s) * HD_ + d] = f2bf(val * SCALE_);
    else if (part == 1) kt[(head * HD_ + d) * S_ + s] = f2bf(val);
    else                v [(head * S_ + s) * HD_ + d] = f2bf(val);
  }
};
struct EpiScores {  // fp32 scores straight into attn_weights region of d_out
  float* c;
  __device__ void operator()(int z, int m, int n, float val) const {
    c[(size_t)z * S_ * S_ + (size_t)m * S_ + n] = val;
  }
};
struct EpiPV {      // attn_out bf16, re-interleaved to [b, s, h*HD+d]
  unsigned short* ao;
  __device__ void operator()(int z, int m, int n, float val) const {
    const int b = z >> 4, h = z & (H_ - 1);
    ao[((size_t)(b * S_ + m)) * E_ + h * HD_ + n] = f2bf(val);
  }
};
struct EpiOut {     // final fp32 output + bias
  float* out; const float* bias;
  __device__ void operator()(int, int m, int n, float val) const {
    out[(size_t)m * E_ + n] = val + bias[n];
  }
};

// ---------- softmax over rows of 2048 (in place, fp32) ----------
__global__ __launch_bounds__(256)
void softmax_rows(float* __restrict__ attn) {
  float* p = attn + (size_t)blockIdx.x * S_;
  const int t = threadIdx.x, lane = t & 31, w = t >> 5;
  __shared__ float redA[8], redB[8];
  float v[8];
  float mx = -3.4e38f;
  #pragma unroll
  for (int i = 0; i < 8; ++i) { v[i] = p[t + i * 256]; mx = fmaxf(mx, v[i]); }
  #pragma unroll
  for (int o = 16; o > 0; o >>= 1) mx = fmaxf(mx, __shfl_xor(mx, o, 32));
  if (lane == 0) redA[w] = mx;
  __syncthreads();
  mx = redA[0];
  #pragma unroll
  for (int j = 1; j < 8; ++j) mx = fmaxf(mx, redA[j]);
  float s = 0.f;
  #pragma unroll
  for (int i = 0; i < 8; ++i) { v[i] = __expf(v[i] - mx); s += v[i]; }
  #pragma unroll
  for (int o = 16; o > 0; o >>= 1) s += __shfl_xor(s, o, 32);
  if (lane == 0) redB[w] = s;
  __syncthreads();
  float tot = 0.f;
  #pragma unroll
  for (int j = 0; j < 8; ++j) tot += redB[j];
  const float inv = 1.0f / tot;
  #pragma unroll
  for (int i = 0; i < 8; ++i) p[t + i * 256] = v[i] * inv;
}

// ---------- prep: fp32 -> bf16 convert / transpose ----------
__global__ void cvt_bf16(const float* __restrict__ s,
                         unsigned short* __restrict__ d, int n) {
  for (int i = blockIdx.x * blockDim.x + threadIdx.x; i < n;
       i += gridDim.x * blockDim.x)
    d[i] = f2bf(s[i]);
}
// s[F][C] -> d[C][F] (writes coalesced)
__global__ void transpose_bf16(const float* __restrict__ s,
                               unsigned short* __restrict__ d, int F, int C) {
  const int total = F * C;
  for (int i = blockIdx.x * blockDim.x + threadIdx.x; i < total;
       i += gridDim.x * blockDim.x) {
    const int e = i / F, f = i - e * F;
    d[i] = f2bf(s[(size_t)f * C + e]);
  }
}

extern "C" void kernel_launch(void* const* d_in, const int* in_sizes, int n_in,
                              void* d_out, int out_size, void* d_ws, size_t ws_size,
                              hipStream_t stream) {
  const float* query = (const float*)d_in[0];
  // d_in[1], d_in[2] (key/value) are ignored by the reference module.
  const float* in_w  = (const float*)d_in[3];
  const float* in_b  = (const float*)d_in[4];
  const float* out_w = (const float*)d_in[5];
  const float* out_b = (const float*)d_in[6];

  float* out  = (float*)d_out;                         // [B,S,E]
  float* attn = out + (size_t)B_ * S_ * E_;            // [B,H,S,S]

  // workspace layout (bf16 as ushort), 48 MiB total
  unsigned short* ws  = (unsigned short*)d_ws;
  size_t o = 0;
  unsigned short* Xb  = ws + o; o += (size_t)B_ * S_ * E_;       // 4M
  unsigned short* Wt  = ws + o; o += (size_t)E_ * 3 * E_;        // 3M  (W^T: [E][3E])
  unsigned short* Wot = ws + o; o += (size_t)E_ * E_;            // 1M  (Wo^T: [E][E])
  unsigned short* q   = ws + o; o += (size_t)B_ * H_ * S_ * HD_; // 4M  [b,h,s,d] (scaled)
  unsigned short* kt  = ws + o; o += (size_t)B_ * H_ * S_ * HD_; // 4M  [b,h,d,s]
  unsigned short* v   = ws + o; o += (size_t)B_ * H_ * S_ * HD_; // 4M  [b,h,s,d]
  unsigned short* ao  = ws + o;                                  // 4M  [b,s,e]

  // 0) precision prep
  cvt_bf16<<<1024, 256, 0, stream>>>(query, Xb, B_ * S_ * E_);
  transpose_bf16<<<1024, 256, 0, stream>>>(in_w, Wt, 3 * E_, E_);
  transpose_bf16<<<1024, 256, 0, stream>>>(out_w, Wot, E_, E_);

  // 1) fused QKV projection: [4096,3072] = Xb[4096,1024] @ Wt[1024,3072] + b
  {
    EpiQKV epi{q, kt, v, in_b};
    dim3 g(3 * E_ / 64, B_ * S_ / 128, 1);
    gemm_wmma<false, EpiQKV><<<g, 256, 0, stream>>>(
        Xb, Wt, B_ * S_, 3 * E_, E_, E_, 3 * E_, 0, 0, epi);
  }
  // 2) scores = q @ k^T per head: [2048,2048] per z, K=64
  {
    EpiScores epi{attn};
    dim3 g(S_ / 64, S_ / 128, B_ * H_);
    gemm_wmma<false, EpiScores><<<g, 256, 0, stream>>>(
        q, kt, S_, S_, HD_, HD_, S_,
        (long long)S_ * HD_, (long long)HD_ * S_, epi);
  }
  // 3) softmax rows, in place in d_out attn region
  softmax_rows<<<B_ * H_ * S_, 256, 0, stream>>>(attn);
  // 4) attn_out = P @ V per head: [2048,64], K=2048; A is fp32 from d_out
  {
    EpiPV epi{ao};
    dim3 g(HD_ / 64, S_ / 128, B_ * H_);
    gemm_wmma<true, EpiPV><<<g, 256, 0, stream>>>(
        attn, v, S_, HD_, S_, S_, HD_,
        (long long)S_ * S_, (long long)S_ * HD_, epi);
  }
  // 5) out = attn_out @ Wo^T + b: [4096,1024]
  {
    EpiOut epi{out, out_b};
    dim3 g(E_ / 64, B_ * S_ / 128, 1);
    gemm_wmma<false, EpiOut><<<g, 256, 0, stream>>>(
        ao, Wot, B_ * S_, E_, E_, E_, E_, 0, 0, epi);
  }
}
